// FrozenHopfield_335007449220
// MI455X (gfx1250) — matrix operand order repaired and checked by reading, not compiled, not run on hardware
//
#include <hip/hip_runtime.h>
#include <hip/hip_bf16.h>

// ---------------- problem constants ----------------
#define B_N   256
#define HW    7056        // 84*84
#define KP    7072        // HW padded to multiple of 32
#define HID   768         // = 24 * 32
#define V_N   50257
#define VP    50304       // V padded to multiple of 128 (chunk size)
#define EPS   1e-8f

typedef __bf16 bh;
typedef __attribute__((ext_vector_type(16))) __bf16 v16bf;
typedef __attribute__((ext_vector_type(8)))  __bf16 v8bf;
typedef __attribute__((ext_vector_type(8)))  float  v8f;

static __device__ __forceinline__ v16bf join8(v8bf lo, v8bf hi) {
  v16bf a;
#pragma unroll
  for (int i = 0; i < 8; ++i) { a[i] = lo[i]; a[i + 8] = hi[i]; }
  return a;
}

// ---------------- workspace layout (bytes), total ~171 MB ----------------
constexpr size_t OFF_OBS  = 0;
constexpr size_t SZ_OBS   = (size_t)B_N * KP * 2;        //  3.62 MB
constexpr size_t OFF_PROJ = OFF_OBS + SZ_OBS;
constexpr size_t SZ_PROJ  = (size_t)HID * KP * 2;        // 10.86 MB
constexpr size_t OFF_WE   = OFF_PROJ + SZ_PROJ;
constexpr size_t SZ_WE    = (size_t)VP * HID * 2;        // 77.27 MB  (row-major [v][h])
constexpr size_t OFF_WET  = OFF_WE + SZ_WE;
constexpr size_t SZ_WET   = (size_t)HID * VP * 2;        // 77.27 MB  (transposed [h][v])
constexpr size_t OFF_QB   = OFF_WET + SZ_WET;
constexpr size_t SZ_QB    = (size_t)B_N * HID * 2;
constexpr size_t OFF_QN   = OFF_QB + SZ_QB;
constexpr size_t SZ_QN    = (size_t)B_N * 4;
constexpr size_t OFF_EN   = OFF_QN + SZ_QN;
constexpr size_t SZ_EN    = (size_t)VP * 4;
constexpr size_t OFF_L    = OFF_EN + SZ_EN;
constexpr size_t SZ_L     = (size_t)B_N * 4;
constexpr size_t OFF_ACC  = OFF_L + SZ_L;
constexpr size_t SZ_ACC   = (size_t)B_N * HID * 4;

// ---------------- K1: channel mean -> bf16, K-padded ----------------
__global__ void k_prep_obs(const float* __restrict__ obs, bh* __restrict__ obsbf) {
  size_t i = (size_t)blockIdx.x * 256 + threadIdx.x;
  if (i >= (size_t)B_N * KP) return;
  int b = (int)(i / KP), k = (int)(i % KP);
  float m = 0.f;
  if (k < HW) {
    const float* p = obs + (size_t)b * 3 * HW + k;
    m = (p[0] + p[HW] + p[2 * HW]) * (1.f / 3.f);
  }
  obsbf[i] = (bh)m;
}

// ---------------- K2: proj -> bf16, K-padded ----------------
__global__ void k_conv_proj(const float* __restrict__ proj, bh* __restrict__ projbf) {
  size_t i = (size_t)blockIdx.x * 256 + threadIdx.x;
  if (i >= (size_t)HID * KP) return;
  int n = (int)(i / KP), k = (int)(i % KP);
  projbf[i] = (bh)(k < HW ? proj[(size_t)n * HW + k] : 0.f);
}

// ---------------- K3: word_embs -> bf16 (V-padded with zeros) + row norms ----------------
__global__ void k_conv_we(const float* __restrict__ we, bh* __restrict__ webf,
                          float* __restrict__ en) {
  __shared__ float red[256];
  int v = blockIdx.x, tid = threadIdx.x;
  float s = 0.f;
  if (v < V_N) {
#pragma unroll
    for (int j = 0; j < 3; ++j) {
      int h = tid + 256 * j;
      float x = we[(size_t)v * HID + h];
      webf[(size_t)v * HID + h] = (bh)x;
      s += x * x;
    }
  } else {
#pragma unroll
    for (int j = 0; j < 3; ++j) webf[(size_t)v * HID + tid + 256 * j] = (bh)0.f;
  }
  red[tid] = s; __syncthreads();
  for (int off = 128; off > 0; off >>= 1) {
    if (tid < off) red[tid] += red[tid + off];
    __syncthreads();
  }
  if (tid == 0) en[v] = (v < V_N) ? sqrtf(red[0]) : 1.0f;
}

// ---------------- K3b: tiled transpose webf[v][h] -> weT[h][v] (64x64 bf16 tiles) ----------------
__global__ void __launch_bounds__(256) k_tr_we(const bh* __restrict__ webf,
                                               bh* __restrict__ weT) {
  __shared__ bh tile[64][72];                 // +8 bf16 pad: bank-clean strided reads
  const int hb = blockIdx.x * 64;             // HID/64 = 12
  const int vb = blockIdx.y * 64;             // VP/64  = 786
  const int t  = threadIdx.x;
#pragma unroll
  for (int j = 0; j < 2; ++j) {               // coalesced read of 64(v) x 64(h)
    int u  = t + 256 * j;                     // 0..511
    int vr = u >> 3;
    int hc = (u & 7) * 8;
    v8bf d = *(const v8bf*)(webf + (size_t)(vb + vr) * HID + hb + hc);
#pragma unroll
    for (int i = 0; i < 8; ++i) tile[vr][hc + i] = d[i];
  }
  __syncthreads();
#pragma unroll
  for (int j = 0; j < 2; ++j) {               // coalesced write of 64(h) x 64(v)
    int u  = t + 256 * j;
    int hr = u >> 3;
    int vc = (u & 7) * 8;
    v8bf d;
#pragma unroll
    for (int i = 0; i < 8; ++i) d[i] = tile[vc + i][hr];
    *(v8bf*)(weT + (size_t)(hb + hr) * VP + vb + vc) = d;
  }
}

// ---------------- K4: q = obs @ proj^T  (bf16 WMMA, one wave / 16x16 tile) ----------------
__global__ void __launch_bounds__(32) k_gemm_q(const bh* __restrict__ obs,
                                               const bh* __restrict__ proj,
                                               bh* __restrict__ qbf) {
  int lane = threadIdx.x, lm = lane & 15, hi = lane >> 4;
  int nbase = blockIdx.x * 16, mbase = blockIdx.y * 16;
  const bh* arow = obs  + (size_t)(mbase + lm) * KP;
  const bh* brow = proj + (size_t)(nbase + lm) * KP;
  v8f c;
#pragma unroll
  for (int r = 0; r < 8; ++r) c[r] = 0.f;
  for (int k0 = 0; k0 < KP; k0 += 32) {
    v16bf a = join8(*(const v8bf*)(arow + k0 + 8 * hi),
                    *(const v8bf*)(arow + k0 + 16 + 8 * hi));
    v16bf b = join8(*(const v8bf*)(brow + k0 + 8 * hi),
                    *(const v8bf*)(brow + k0 + 16 + 8 * hi));
    c = __builtin_amdgcn_wmma_f32_16x16x32_bf16(false, a, false, b, (short)0, c, false, false);
  }
#pragma unroll
  for (int r = 0; r < 8; ++r)
    qbf[(size_t)(mbase + r + 8 * hi) * HID + nbase + lm] = (bh)c[r];
}

// ---------------- K5: q row norms ----------------
__global__ void k_qnorm(const bh* __restrict__ qbf, float* __restrict__ qn) {
  __shared__ float red[256];
  int b = blockIdx.x, tid = threadIdx.x;
  float s = 0.f;
#pragma unroll
  for (int j = 0; j < 3; ++j) {
    float x = (float)qbf[(size_t)b * HID + tid + 256 * j];
    s += x * x;
  }
  red[tid] = s; __syncthreads();
  for (int off = 128; off > 0; off >>= 1) {
    if (tid < off) red[tid] += red[tid + off];
    __syncthreads();
  }
  if (tid == 0) qn[b] = sqrtf(red[0]);
}

// ---------------- K6: zero accumulators (re-zeroed every call) ----------------
__global__ void k_zero(float* __restrict__ acc, float* __restrict__ lacc) {
  int i = blockIdx.x * 256 + threadIdx.x;
  if (i < B_N * HID) acc[i] = 0.f;
  if (i < B_N) lacc[i] = 0.f;
}

// ---------------- K7: fused sims -> exp -> (P @ We) partials, all-wave-balanced ----------------
// grid = (57 V-slices, 16 q-blocks), block = 256 = 8 waves; chunk = 128 vocab rows.
// Wave w: (a) S tile for vocab cols [v0+16w, v0+16w+16) — 24 WMMAs over K=768,
//         (b) P@V rank-128 update on its 96-wide HID slice — 24 WMMAs.
// |beta * cos_sim| <= 8 -> exp is overflow-safe, no max pass needed.
__global__ void __launch_bounds__(256) k_attn(const bh* __restrict__ qbf,
                                              const bh* __restrict__ webf,
                                              const bh* __restrict__ weT,
                                              const float* __restrict__ qn,
                                              const float* __restrict__ en,
                                              const int* __restrict__ betap,
                                              float* __restrict__ acc,
                                              float* __restrict__ lacc) {
  __shared__ bh    p_lds[16][136];   // P[16 x 128] bridge (C-layout -> A-layout), 272B stride
  __shared__ float l_s[16];          // per-q-row exp sums for this slice

  const int tid = threadIdx.x, lane = tid & 31, wave = tid >> 5;
  const int lm = lane & 15, hi = lane >> 4;
  const int mbase = blockIdx.y * 16;
  const int NCH = VP / 128;          // 393 chunks
  const int CPB = 7;                 // ceil(393/57)
  int c0 = blockIdx.x * CPB, c1 = min(c0 + CPB, NCH);
  if (c0 >= c1) return;

  const float betaf = (float)betap[0];
  float qn_r[8];
#pragma unroll
  for (int r = 0; r < 8; ++r) qn_r[r] = qn[mbase + r + 8 * hi];
  if (tid < 16) l_s[tid] = 0.f;

  v8f cacc[6];
#pragma unroll
  for (int t = 0; t < 6; ++t) {
#pragma unroll
    for (int r = 0; r < 8; ++r) cacc[t][r] = 0.f;
  }
  const int h0w = wave * 96;                       // this wave's HID slice
  const bh* arow = qbf + (size_t)(mbase + lm) * HID;

  for (int ch = c0; ch < c1; ++ch) {
    const int v0 = ch * 128;
    __syncthreads();                               // protect p_lds reuse across chunks

    // ---- (a) this wave's S tile: 16 q-rows x 16 vocab cols ----
    const int vcol = v0 + 16 * wave + lm;          // this lane's vocab column
    v8f s;
#pragma unroll
    for (int r = 0; r < 8; ++r) s[r] = 0.f;
    const bh* brow = webf + (size_t)vcol * HID;
    for (int k0 = 0; k0 < HID; k0 += 32) {
      v16bf a = join8(*(const v8bf*)(arow + k0 + 8 * hi),
                      *(const v8bf*)(arow + k0 + 16 + 8 * hi));
      v16bf b = join8(*(const v8bf*)(brow + k0 + 8 * hi),
                      *(const v8bf*)(brow + k0 + 16 + 8 * hi));
      s = __builtin_amdgcn_wmma_f32_16x16x32_bf16(false, a, false, b, (short)0, s, false, false);
    }
    const float env = en[vcol];
    const bool valid = (vcol < V_N);
#pragma unroll
    for (int r = 0; r < 8; ++r) {
      float den = qn_r[r] * env + EPS;
      float p = valid ? __expf(betaf * s[r] / den) : 0.f;
      float rs = p;                                // reduce across the 16 lanes of this row
      rs += __shfl_xor(rs, 1, 32);
      rs += __shfl_xor(rs, 2, 32);
      rs += __shfl_xor(rs, 4, 32);
      rs += __shfl_xor(rs, 8, 32);
      if (lm == 0) atomicAdd(&l_s[r + 8 * hi], rs);
      p_lds[r + 8 * hi][16 * wave + lm] = (bh)p;   // C-layout -> row-major P
    }
    __syncthreads();

    // ---- (b) acc[16, 96] += P[16,128] @ We[128, 96], B-frags from transposed copy ----
#pragma unroll
    for (int ks = 0; ks < 4; ++ks) {
      v16bf ap = join8(*(const v8bf*)(&p_lds[lm][32 * ks + 8 * hi]),
                       *(const v8bf*)(&p_lds[lm][32 * ks + 16 + 8 * hi]));
#pragma unroll
      for (int ht = 0; ht < 6; ++ht) {
        int h = h0w + 16 * ht + lm;
        v16bf bw = join8(*(const v8bf*)(weT + (size_t)h * VP + v0 + 32 * ks + 8 * hi),
                         *(const v8bf*)(weT + (size_t)h * VP + v0 + 32 * ks + 16 + 8 * hi));
        cacc[ht] = __builtin_amdgcn_wmma_f32_16x16x32_bf16(false, ap, false, bw, (short)0,
                                                           cacc[ht], false, false);
      }
    }
    if (ch + 1 < c1)                               // hint next chunk toward L2/L0
      __builtin_prefetch((const void*)(webf + (size_t)(v0 + 128) * HID + tid * 384), 0, 1);
  }

  // combine slice partials
#pragma unroll
  for (int ht = 0; ht < 6; ++ht) {
    int h = h0w + 16 * ht + lm;
#pragma unroll
    for (int r = 0; r < 8; ++r)
      atomicAdd(&acc[(size_t)(mbase + r + 8 * hi) * HID + h], cacc[ht][r]);
  }
  __syncthreads();
  if (tid < 16) atomicAdd(&lacc[mbase + tid], l_s[tid]);
}

// ---------------- K8: normalize ----------------
__global__ void k_fin(const float* __restrict__ acc, const float* __restrict__ lacc,
                      float* __restrict__ out) {
  int i = blockIdx.x * 256 + threadIdx.x;
  if (i < B_N * HID) out[i] = acc[i] / lacc[i / HID];
}

// ---------------- launcher ----------------
extern "C" void kernel_launch(void* const* d_in, const int* in_sizes, int n_in,
                              void* d_out, int out_size, void* d_ws, size_t ws_size,
                              hipStream_t stream) {
  (void)in_sizes; (void)n_in; (void)out_size; (void)ws_size; // needs ~171 MB of d_ws
  const float* obs  = (const float*)d_in[0];
  const float* we   = (const float*)d_in[1];
  const float* proj = (const float*)d_in[2];
  const int*   beta = (const int*)d_in[3];

  char* ws = (char*)d_ws;
  bh*    obsbf  = (bh*)(ws + OFF_OBS);
  bh*    projbf = (bh*)(ws + OFF_PROJ);
  bh*    webf   = (bh*)(ws + OFF_WE);
  bh*    weT    = (bh*)(ws + OFF_WET);
  bh*    qbf    = (bh*)(ws + OFF_QB);
  float* qn     = (float*)(ws + OFF_QN);
  float* en     = (float*)(ws + OFF_EN);
  float* lacc   = (float*)(ws + OFF_L);
  float* acc    = (float*)(ws + OFF_ACC);

  k_prep_obs <<<(B_N * KP + 255) / 256, 256, 0, stream>>>(obs, obsbf);
  k_conv_proj<<<(HID * KP + 255) / 256, 256, 0, stream>>>(proj, projbf);
  k_conv_we  <<<VP, 256, 0, stream>>>(we, webf, en);
  k_tr_we    <<<dim3(HID / 64, VP / 64), 256, 0, stream>>>(webf, weT);
  k_gemm_q   <<<dim3(HID / 16, B_N / 16), 32, 0, stream>>>(obsbf, projbf, qbf);
  k_qnorm    <<<B_N, 256, 0, stream>>>(qbf, qn);
  k_zero     <<<(B_N * HID + 255) / 256, 256, 0, stream>>>(acc, lacc);
  k_attn     <<<dim3(57, 16), 256, 0, stream>>>(qbf, webf, weT, qn, en, beta, acc, lacc);
  k_fin      <<<(B_N * HID + 255) / 256, 256, 0, stream>>>(acc, lacc, (float*)d_out);
}
